// PANLoss_17111149707829
// MI455X (gfx1250) — compile-verified, matrix operand order checked by reading, not run on hardware
//
#include <hip/hip_runtime.h>
#include <hip/hip_bf16.h>

#define BATCH 8
#define HH 640
#define WW 640
#define HW (HH*WW)          // 409600
#define CCH 4
#define SP 16               // padded segment count (9 used)
#define NB 1024             // histogram bins per level
#define CHUNK 4096
#define BPB (HW/CHUNK)      // 100 blocks per batch image

typedef float v2f __attribute__((ext_vector_type(2)));
typedef float v8f __attribute__((ext_vector_type(8)));
typedef _Float16 h16 __attribute__((ext_vector_type(16)));

#if __has_builtin(__builtin_amdgcn_wmma_f32_16x16x4_f32)
#define HAVE_WMMA_F32X4 1
#elif __has_builtin(__builtin_amdgcn_wmma_f32_16x16x32_f16)
#define HAVE_WMMA_F16 1
#endif

struct WS {
    float ak[BATCH], bk[BATCH], ck[BATCH];     // kernel dice partials
    float at[BATCH], bt[BATCH], ct[BATCH];     // text dice partials
    unsigned pos_num[BATCH], neg_cnt[BATCH];
    unsigned hist0[BATCH][NB];
    unsigned hist1[BATCH][NB];
    float thr_lo[BATCH], thr_hi[BATCH];
    unsigned kk[BATCH];
    float thr[BATCH];
    float cnt_k[BATCH][SP], cnt_t[BATCH][SP];
    float sum_k[BATCH][SP][CCH];
    float sum_l[BATCH][SP];
    float Kmean[BATCH][SP][CCH];
    float Ldis[BATCH];
};

__global__ void k_zero(unsigned* p, size_t n) {
    size_t i = (size_t)blockIdx.x * blockDim.x + threadIdx.x;
    size_t stride = (size_t)gridDim.x * blockDim.x;
    for (; i < n; i += stride) p[i] = 0u;
}

__device__ __forceinline__ float blockReduce256(float v, float* sbuf) {
    int t = threadIdx.x;
    __syncthreads();
    sbuf[t] = v;
    __syncthreads();
    for (int s = 128; s > 0; s >>= 1) {
        if (t < s) sbuf[t] += sbuf[t + s];
        __syncthreads();
    }
    return sbuf[0];
}

__device__ __forceinline__ float dsigmoid(float x) {
    return 1.0f / (1.0f + __expf(-x));
}

// ---------------- Pass 1: sigmoids, kernel dice, OHEM hist L0, segment sums
__global__ __launch_bounds__(256)
void k_main(const float* __restrict__ tl, const float* __restrict__ kl,
            const float* __restrict__ sim, const float* __restrict__ mask,
            const int* __restrict__ tkey, const int* __restrict__ kkey,
            WS* ws) {
    int b = blockIdx.x / BPB;
    int chunk = blockIdx.x % BPB;
    size_t pix0 = (size_t)chunk * CHUNK;              // pixel offset in image
    size_t base = (size_t)b * HW + pix0;              // flat offset in [B,H,W]
    int t = threadIdx.x;

    __shared__ unsigned h[NB];
    __shared__ float s_cnt_k[SP], s_cnt_t[SP], s_sum_k[SP * CCH];
    __shared__ float sbuf[256];
    for (int i = t; i < NB; i += 256) h[i] = 0u;
    if (t < SP) { s_cnt_k[t] = 0.f; s_cnt_t[t] = 0.f; }
    for (int i = t; i < SP * CCH; i += 256) s_sum_k[i] = 0.f;
    __syncthreads();

    float ak = 0.f, bk = 0.f, ck = 0.f, posn = 0.f, negn = 0.f;
    for (int i = t; i < CHUNK; i += 256) {
        size_t idx = base + i;
        __builtin_prefetch((const void*)(tl + idx + 1024), 0, 0);
        float xt = tl[idx];
        float xk = kl[idx];
        float m  = mask[idx];
        int tk = tkey[idx];
        int kv = kkey[idx];
        bool tm = m > 0.5f;
        float pt = dsigmoid(xt);
        float pk = dsigmoid(xk);
        bool selt = xt > 0.f;
        bool selk = xk > 0.f;
        bool gtp = tk > 0;
        bool gkp = kv > 0;
        if (selt && tm) {               // km mask (detached)
            float g = gkp ? 1.f : 0.f;
            ak += pk * g;
            bk += pk * pk;
            ck += g;
        }
        if (gtp) {
            if (tm) posn += 1.f;
        } else {
            negn += 1.f;
            int bin = (int)(pt * (float)NB);
            bin = bin < 0 ? 0 : (bin > NB - 1 ? NB - 1 : bin);
            atomicAdd(&h[bin], 1u);
        }
        int lt = (selt && tm) ? tk : 0;
        int lk = (selk && tm) ? kv : 0;
        if (lt > 0) atomicAdd(&s_cnt_t[lt], 1.f);
        if (lk > 0) {
            atomicAdd(&s_cnt_k[lk], 1.f);
            size_t sb = (size_t)b * CCH * HW + pix0 + i;
            #pragma unroll
            for (int c = 0; c < CCH; ++c)
                atomicAdd(&s_sum_k[lk * CCH + c], sim[sb + (size_t)c * HW]);
        }
    }

    float r;
    r = blockReduce256(ak, sbuf);   if (t == 0) atomicAdd(&ws->ak[b], r);
    r = blockReduce256(bk, sbuf);   if (t == 0) atomicAdd(&ws->bk[b], r);
    r = blockReduce256(ck, sbuf);   if (t == 0) atomicAdd(&ws->ck[b], r);
    r = blockReduce256(posn, sbuf); if (t == 0) atomicAdd(&ws->pos_num[b], (unsigned)(r + 0.5f));
    r = blockReduce256(negn, sbuf); if (t == 0) atomicAdd(&ws->neg_cnt[b], (unsigned)(r + 0.5f));
    __syncthreads();
    for (int i = t; i < NB; i += 256)
        if (h[i]) atomicAdd(&ws->hist0[b][i], h[i]);
    if (t < SP) {
        if (s_cnt_k[t] != 0.f) atomicAdd(&ws->cnt_k[b][t], s_cnt_k[t]);
        if (s_cnt_t[t] != 0.f) atomicAdd(&ws->cnt_t[b][t], s_cnt_t[t]);
    }
    float* gsk = &ws->sum_k[b][0][0];
    for (int i = t; i < SP * CCH; i += 256)
        if (s_sum_k[i] != 0.f) atomicAdd(&gsk[i], s_sum_k[i]);
}

// ---------------- kernel means + OHEM scan level 0
__global__ void k_scan1(WS* ws) {
    int b = blockIdx.x;
    int t = threadIdx.x;
    for (int i = t; i < SP * CCH; i += blockDim.x) {
        int s = i / CCH, c = i % CCH;
        ws->Kmean[b][s][c] = ws->sum_k[b][s][c] / fmaxf(ws->cnt_k[b][s], 1.f);
    }
    if (t == 0) {
        unsigned pos = ws->pos_num[b];
        unsigned neg = ws->neg_cnt[b];
        unsigned k = (3u * pos < neg) ? 3u * pos : neg;
        if (pos == 0u || k == 0u) {
            ws->thr[b] = -1.f;          // fallback: select == mask
            ws->thr_lo[b] = -2.f;
            ws->thr_hi[b] = -1.f;
            ws->kk[b] = 1u;
        } else {
            unsigned cum = 0; int sel = 0; unsigned rem = 1u;
            for (int bin = NB - 1; bin >= 0; --bin) {
                unsigned c = ws->hist0[b][bin];
                if (cum + c >= k) { sel = bin; rem = k - cum; break; }
                cum += c;
            }
            ws->thr_lo[b] = (float)sel * (1.f / NB);
            ws->thr_hi[b] = (float)(sel + 1) * (1.f / NB);
            ws->kk[b] = rem;
        }
    }
}

// ---------------- OHEM refine (level-1 histogram inside bracket)
__global__ __launch_bounds__(256)
void k_refine(const float* __restrict__ tl, const int* __restrict__ tkey, WS* ws) {
    int b = blockIdx.x / BPB;
    int chunk = blockIdx.x % BPB;
    float lo = ws->thr_lo[b], hi = ws->thr_hi[b];
    if (lo < 0.f) return;               // fallback image: nothing to refine
    size_t base = (size_t)b * HW + (size_t)chunk * CHUNK;
    int t = threadIdx.x;
    __shared__ unsigned h[NB];
    for (int i = t; i < NB; i += 256) h[i] = 0u;
    __syncthreads();
    float inv = (float)NB / (hi - lo);
    for (int i = t; i < CHUNK; i += 256) {
        size_t idx = base + i;
        if (tkey[idx] > 0) continue;    // negatives only
        float pt = dsigmoid(tl[idx]);
        if (pt >= lo && pt < hi) {
            int bin = (int)((pt - lo) * inv);
            bin = bin < 0 ? 0 : (bin > NB - 1 ? NB - 1 : bin);
            atomicAdd(&h[bin], 1u);
        }
    }
    __syncthreads();
    for (int i = t; i < NB; i += 256)
        if (h[i]) atomicAdd(&ws->hist1[b][i], h[i]);
}

__global__ void k_scan2(WS* ws) {
    int b = blockIdx.x;
    if (threadIdx.x != 0) return;
    float lo = ws->thr_lo[b], hi = ws->thr_hi[b];
    if (lo < 0.f) { ws->thr[b] = -1.f; return; }
    unsigned k = ws->kk[b], cum = 0;
    float thr = lo;
    for (int bin = NB - 1; bin >= 0; --bin) {
        unsigned c = ws->hist1[b][bin];
        if (cum + c >= k) { thr = lo + (float)bin * (hi - lo) * (1.f / NB); break; }
        cum += c;
    }
    ws->thr[b] = thr;
}

// ---------------- discrimination loss: Gram of kernel means via WMMA f32 16x16x4
__global__ void k_dis(WS* ws) {
    int b = blockIdx.x;
    int lane = threadIdx.x;       // 32 threads = 1 wave
    __shared__ float G[16][16];
    const float* Km = &ws->Kmean[b][0][0];   // [16][4]
    int r = lane & 15;

#if defined(HAVE_WMMA_F32X4)
    // A 16x4 f32 layout: lanes 0-15 hold (K0,K1), lanes 16-31 hold (K2,K3).
    // B = A^T (4x16) has the identical register image.
    v2f av;
    if (lane < 16) { av.x = Km[r * 4 + 0]; av.y = Km[r * 4 + 1]; }
    else           { av.x = Km[r * 4 + 2]; av.y = Km[r * 4 + 3]; }
    v8f acc = {};
    v8f g = __builtin_amdgcn_wmma_f32_16x16x4_f32(
        false, av, false, av, (short)0, acc, false, false);
    int mbase = (lane < 16) ? 0 : 8;
    #pragma unroll
    for (int j = 0; j < 8; ++j) G[mbase + j][r] = g[j];
#elif defined(HAVE_WMMA_F16)
    // 16-bit A 16x32: lanes 0-15 VGPR0-1 hold K=0..3 -> elements 0..3; rest 0.
    h16 ah;
    #pragma unroll
    for (int e = 0; e < 16; ++e) ah[e] = (_Float16)0.f;
    if (lane < 16) {
        ah[0] = (_Float16)Km[r * 4 + 0];
        ah[1] = (_Float16)Km[r * 4 + 1];
        ah[2] = (_Float16)Km[r * 4 + 2];
        ah[3] = (_Float16)Km[r * 4 + 3];
    }
    v8f acc = {};
    v8f g = __builtin_amdgcn_wmma_f32_16x16x32_f16(
        false, ah, false, ah, (short)0, acc, false, false);
    int mbase = (lane < 16) ? 0 : 8;
    #pragma unroll
    for (int j = 0; j < 8; ++j) G[mbase + j][r] = g[j];
#else
    for (int idx = lane; idx < 256; idx += 32) {
        int i = idx >> 4, j = idx & 15;
        float d = 0.f;
        #pragma unroll
        for (int c = 0; c < CCH; ++c) d += Km[i * 4 + c] * Km[j * 4 + c];
        G[i][j] = d;
    }
#endif
    __syncthreads();
    if (lane == 0) {
        float sum = 0.f; int np = 0;
        for (int i = 1; i < 9; ++i) {
            if (ws->cnt_k[b][i] <= 0.f) continue;
            for (int j = 1; j < 9; ++j) {
                if (j == i || ws->cnt_k[b][j] <= 0.f) continue;
                float d2 = G[i][i] + G[j][j] - 2.f * G[i][j];
                float d = sqrtf(fmaxf(d2, 0.f));
                float q = fmaxf(3.0f - d, 0.f);
                sum += __logf(q * q + 1.f);
                np++;
            }
        }
        ws->Ldis[b] = sum / (float)(np > 0 ? np : 1);
    }
}

// ---------------- Pass 3: OHEM-masked text dice + aggregation loss
__global__ __launch_bounds__(256)
void k_agg(const float* __restrict__ tl, const float* __restrict__ sim,
           const float* __restrict__ mask, const int* __restrict__ tkey,
           WS* ws) {
    int b = blockIdx.x / BPB;
    int chunk = blockIdx.x % BPB;
    size_t pix0 = (size_t)chunk * CHUNK;
    size_t base = (size_t)b * HW + pix0;
    int t = threadIdx.x;

    __shared__ float Kl[SP * CCH];
    __shared__ float s_sum_l[SP];
    __shared__ float sbuf[256];
    if (t < SP * CCH) Kl[t] = (&ws->Kmean[b][0][0])[t];
    if (t < SP) s_sum_l[t] = 0.f;
    __syncthreads();

    float thr = ws->thr[b];
    float at = 0.f, bt = 0.f, ct = 0.f;
    for (int i = t; i < CHUNK; i += 256) {
        size_t idx = base + i;
        __builtin_prefetch((const void*)(tl + idx + 1024), 0, 0);
        float xt = tl[idx];
        float m  = mask[idx];
        int tk = tkey[idx];
        bool tm = m > 0.5f;
        float pt = dsigmoid(xt);
        bool pos = tk > 0;
        float om = (((pt >= thr) || pos) && tm) ? 1.f : 0.f;
        float g = pos ? 1.f : 0.f;
        at += pt * g * om;
        bt += pt * pt * om;
        ct += g * om;
        int lt = (xt > 0.f && tm) ? tk : 0;
        if (lt > 0) {
            size_t sb = (size_t)b * CCH * HW + pix0 + i;
            float d2 = 0.f;
            #pragma unroll
            for (int c = 0; c < CCH; ++c) {
                float df = sim[sb + (size_t)c * HW] - Kl[lt * CCH + c];
                d2 += df * df;
            }
            float d = sqrtf(d2);
            float q = fmaxf(d - 0.5f, 0.f);
            atomicAdd(&s_sum_l[lt], __logf(q * q + 1.f));
        }
    }

    float r;
    r = blockReduce256(at, sbuf); if (t == 0) atomicAdd(&ws->at[b], r);
    r = blockReduce256(bt, sbuf); if (t == 0) atomicAdd(&ws->bt[b], r);
    r = blockReduce256(ct, sbuf); if (t == 0) atomicAdd(&ws->ct[b], r);
    __syncthreads();
    if (t < SP && s_sum_l[t] != 0.f) atomicAdd(&ws->sum_l[b][t], s_sum_l[t]);
}

// ---------------- finalize
__global__ void k_final(WS* ws, float* out) {
    if (threadIdx.x != 0 || blockIdx.x != 0) return;
    const float eps = 1e-6f;
    float lt = 0.f, lk = 0.f, la = 0.f, ld = 0.f;
    for (int b = 0; b < BATCH; ++b) {
        lk += 1.f - 2.f * ws->ak[b] / (ws->bk[b] + ws->ck[b] + 2.f * eps);
        lt += 1.f - 2.f * ws->at[b] / (ws->bt[b] + ws->ct[b] + 2.f * eps);
        float s = 0.f; int nv = 0;
        for (int seg = 1; seg < 9; ++seg) {
            if (ws->cnt_t[b][seg] > 0.f && ws->cnt_k[b][seg] > 0.f) {
                s += ws->sum_l[b][seg] / fmaxf(ws->cnt_t[b][seg], 1.f);
                nv++;
            }
        }
        la += s / (float)(nv > 0 ? nv : 1);
        ld += ws->Ldis[b];
    }
    float invB = 1.f / (float)BATCH;
    out[0] = lt * invB + 0.5f * (lk * invB) + 0.25f * (la * invB + ld * invB);
}

extern "C" void kernel_launch(void* const* d_in, const int* in_sizes, int n_in,
                              void* d_out, int out_size, void* d_ws, size_t ws_size,
                              hipStream_t stream) {
    const float* tl   = (const float*)d_in[0];   // pre_text_logits [B,H,W]
    const float* kl   = (const float*)d_in[1];   // pre_kernel_logits
    const float* sim  = (const float*)d_in[2];   // similarity [B,C,H,W]
    // d_in[3]=gt_text, d_in[4]=gt_kernel derived from keys; not read
    const float* mask = (const float*)d_in[5];   // train_mask
    const int*  tkey  = (const int*)d_in[6];     // gt_text_key
    const int*  kkey  = (const int*)d_in[7];     // gt_kernel_key
    float* out = (float*)d_out;
    WS* ws = (WS*)d_ws;

    size_t nwords = (sizeof(WS) + 3) / 4;
    k_zero<<<(unsigned)((nwords + 255) / 256), 256, 0, stream>>>((unsigned*)d_ws, nwords);

    dim3 gmain(BATCH * BPB);
    k_main<<<gmain, 256, 0, stream>>>(tl, kl, sim, mask, tkey, kkey, ws);
    k_scan1<<<BATCH, 64, 0, stream>>>(ws);
    k_refine<<<gmain, 256, 0, stream>>>(tl, tkey, ws);
    k_scan2<<<BATCH, 32, 0, stream>>>(ws);
    k_dis<<<BATCH, 32, 0, stream>>>(ws);
    k_agg<<<gmain, 256, 0, stream>>>(tl, sim, mask, tkey, ws);
    k_final<<<1, 32, 0, stream>>>(ws, out);
}
